// Block_3058016715037
// MI455X (gfx1250) — matrix operand order repaired and checked by reading, not compiled, not run on hardware
//
#include <hip/hip_runtime.h>
#include <hip/hip_bf16.h>
#include <math.h>

typedef __attribute__((ext_vector_type(16))) _Float16 v16h;
typedef __attribute__((ext_vector_type(8)))  float    v8f;

#define DIMC   768
#define HEADS  12
#define HD     64
#define HIDDEN 3072
#define BATCH  4
#define SEQ    2048
#define ROWS   (BATCH*SEQ)   // 8192

static __device__ __forceinline__ v8f vzero8() {
  v8f z = {0.f,0.f,0.f,0.f,0.f,0.f,0.f,0.f};
  return z;
}

// ---- CDNA5 async global->LDS copy (ASYNCcnt-tracked, bypasses VGPRs) ----
// LDS address is taken from the VGPR operand; low 32 bits of a flat-cast LDS pointer are
// the workgroup-relative LDS offset (flat->LDS aperture truncation).
static __device__ __forceinline__ void async_copy_b128(void* lds_dst, const void* gsrc) {
  uint32_t l = (uint32_t)(uintptr_t)lds_dst;
  uint64_t g = (uint64_t)(uintptr_t)gsrc;
  asm volatile("global_load_async_to_lds_b128 %0, %1, off"
               :: "v"(l), "v"(g) : "memory");
}
static __device__ __forceinline__ void async_wait0() {
  asm volatile("s_wait_asynccnt 0x0" ::: "memory");
}

// ---- WMMA fragment loaders (contiguous k = 16*g + s order, same for A and B; wave32) ----
// A (16x32 f16): lane m = lane&15, group g = lane>>4.
// vgpr v holds K = 2*v + 16*g  (pair K, K+1)
static __device__ __forceinline__ v16h load_a_frag(const _Float16* p, int ld, int lane) {
  const int m = lane & 15, g = lane >> 4;
  v16h a;
#pragma unroll
  for (int v = 0; v < 8; ++v) {
    const int k = 2*v + 16*g;
    a[2*v]   = p[m*ld + k];
    a[2*v+1] = p[m*ld + k + 1];
  }
  return a;
}

// B (32x16 f16) from K-transposed storage: element B[k][n] at p[n*ld + k].
// vgpr v holds K = 2*v + 16*g (pair K, K+1), n = lane&15.
static __device__ __forceinline__ v16h load_b_frag_t(const _Float16* p, int ld, int lane) {
  const int n = lane & 15, g = lane >> 4;
  v16h b;
#pragma unroll
  for (int v = 0; v < 8; ++v) {
    const int k = 2*v + 16*g;
    b[2*v]   = p[n*ld + k];
    b[2*v+1] = p[n*ld + k + 1];
  }
  return b;
}

// B (32x16 f16) from row-major storage: element B[k][n] at p[k*ld + n].
static __device__ __forceinline__ v16h load_b_frag_rm(const _Float16* p, int ld, int lane) {
  const int n = lane & 15, g = lane >> 4;
  v16h b;
#pragma unroll
  for (int v = 0; v < 8; ++v) {
    const int k = 2*v + 16*g;
    b[2*v]   = p[(k)  *ld + n];
    b[2*v+1] = p[(k+1)*ld + n];
  }
  return b;
}

static __device__ __forceinline__ v8f wmma_f16(v16h a, v16h b, v8f c) {
  return __builtin_amdgcn_wmma_f32_16x16x32_f16(false, a, false, b, (short)0, c, false, false);
}

// ---------------- weight conversion f32 -> f16, transposed [K][N] -> [N][K] --
__global__ void cvt_t_kernel(const float* __restrict__ s, _Float16* __restrict__ d,
                             int K, int Nn) {
  int i = blockIdx.x * 256 + threadIdx.x;
  if (i < K * Nn) {
    int k = i / Nn, n = i - k * Nn;
    d[(size_t)n * K + k] = (_Float16)s[i];
  }
}

// ---------------- LayerNorm (row of 768) -> f16 ----------------
__global__ __launch_bounds__(256) void ln_kernel(const float* __restrict__ x,
                                                 const float* __restrict__ w,
                                                 const float* __restrict__ b,
                                                 _Float16* __restrict__ out) {
  __shared__ float sred[256];
  __shared__ float smu, srs;
  const int row = blockIdx.x, t = threadIdx.x;
  const float* xr = x + (size_t)row * DIMC;
  float v0 = xr[t], v1 = xr[t + 256], v2 = xr[t + 512];
  sred[t] = v0 + v1 + v2;
  __syncthreads();
  for (int o = 128; o > 0; o >>= 1) { if (t < o) sred[t] += sred[t + o]; __syncthreads(); }
  if (t == 0) smu = sred[0] * (1.0f / DIMC);
  __syncthreads();
  const float mu = smu;
  float d0 = v0 - mu, d1 = v1 - mu, d2 = v2 - mu;
  sred[t] = d0*d0 + d1*d1 + d2*d2;
  __syncthreads();
  for (int o = 128; o > 0; o >>= 1) { if (t < o) sred[t] += sred[t + o]; __syncthreads(); }
  if (t == 0) srs = rsqrtf(sred[0] * (1.0f / DIMC) + 1e-5f);
  __syncthreads();
  const float rs = srs;
  _Float16* orow = out + (size_t)row * DIMC;
  orow[t]       = (_Float16)(d0 * rs * w[t]       + b[t]);
  orow[t + 256] = (_Float16)(d1 * rs * w[t + 256] + b[t + 256]);
  orow[t + 512] = (_Float16)(d2 * rs * w[t + 512] + b[t + 512]);
}

// ---------------- Generic WMMA GEMM: C(MxN) = A(MxK) * Wt(NxK)^T + epilogue --
// Block tile 128x128, 8 waves (4 row x 2 col), wave tile 32x64, K-step 32.
// A and Wt both staged [row][k] via async b128 copies (contiguous k-pairs).
// EPI 0: bias + split/store f16 q,k,v per-head layout [row][head][64]
// EPI 1: bias + out_f32 = resid + val*gamma   (proj / fc2)
// EPI 2: bias + exact GELU -> f16             (fc1)
template<int EPI>
__global__ __launch_bounds__(256) void gemm_kernel(const _Float16* __restrict__ A,
                                                   const _Float16* __restrict__ Wt,
                                                   const float* __restrict__ bias,
                                                   int M, int Nn, int K,
                                                   const float* __restrict__ resid,
                                                   const float* __restrict__ gamma,
                                                   float* __restrict__ outf,
                                                   _Float16* __restrict__ outh,
                                                   _Float16* __restrict__ qo,
                                                   _Float16* __restrict__ ko,
                                                   _Float16* __restrict__ vo) {
  __shared__ __align__(16) _Float16 As[128 * 32];   // [m][k]
  __shared__ __align__(16) _Float16 Bs[128 * 32];   // [n][k]

  const int tid  = threadIdx.x;
  const int lane = tid & 31;
  const int wv   = tid >> 5;
  const int wr   = wv & 3, wc = wv >> 2;
  const int m0 = blockIdx.y * 128;
  const int n0 = blockIdx.x * 128;

  // staging addresses: 2 threads per row, 16 halves (32B) each = 2x b128
  const int srow = tid >> 1, skoff = (tid & 1) * 16;
  const _Float16* gA = A  + (size_t)(m0 + srow) * K + skoff;
  const _Float16* gB = Wt + (size_t)(n0 + srow) * K + skoff;
  _Float16* lA = As + srow * 32 + skoff;
  _Float16* lB = Bs + srow * 32 + skoff;

  v8f acc[2][4];
#pragma unroll
  for (int i = 0; i < 2; ++i)
#pragma unroll
    for (int j = 0; j < 4; ++j) acc[i][j] = vzero8();

  for (int k0 = 0; k0 < K; k0 += 32) {
    async_copy_b128(lA,      gA + k0);
    async_copy_b128(lA + 8,  gA + k0 + 8);
    async_copy_b128(lB,      gB + k0);
    async_copy_b128(lB + 8,  gB + k0 + 8);
    if (k0 + 32 < K) {                       // prefetch next K-chunk
      __builtin_prefetch(gA + k0 + 32, 0, 0);
      __builtin_prefetch(gB + k0 + 32, 0, 0);
    }
    async_wait0();
    __syncthreads();

    v16h af[2], bf[4];
#pragma unroll
    for (int i = 0; i < 2; ++i) af[i] = load_a_frag(As + (wr * 32 + i * 16) * 32, 32, lane);
#pragma unroll
    for (int j = 0; j < 4; ++j) bf[j] = load_b_frag_t(Bs + (wc * 64 + j * 16) * 32, 32, lane);
#pragma unroll
    for (int i = 0; i < 2; ++i)
#pragma unroll
      for (int j = 0; j < 4; ++j) acc[i][j] = wmma_f16(af[i], bf[j], acc[i][j]);
    __syncthreads();
  }

  // epilogue: C layout row = r + 8*g, col = lane&15
  const int n15 = lane & 15, g = lane >> 4;
#pragma unroll
  for (int i = 0; i < 2; ++i)
#pragma unroll
    for (int j = 0; j < 4; ++j)
#pragma unroll
      for (int r = 0; r < 8; ++r) {
        const int row = m0 + wr * 32 + i * 16 + r + 8 * g;
        const int col = n0 + wc * 64 + j * 16 + n15;
        float val = acc[i][j][r] + bias[col];
        if constexpr (EPI == 0) {
          const int which = col / DIMC;
          const int hh = (col % DIMC) / HD;
          const int dd = col % HD;
          _Float16* dst = (which == 0) ? qo : (which == 1) ? ko : vo;
          dst[((size_t)row * HEADS + hh) * HD + dd] = (_Float16)val;
        } else if constexpr (EPI == 1) {
          outf[(size_t)row * Nn + col] = resid[(size_t)row * Nn + col] + val * gamma[col];
        } else {
          const float gv = 0.5f * val * (1.0f + erff(val * 0.70710678118654752f));
          outh[(size_t)row * Nn + col] = (_Float16)gv;
        }
      }
}

// ---------------- Flash attention: block = (b,h) x 128 query rows ----------
__global__ __launch_bounds__(256) void attn_kernel(const _Float16* __restrict__ q,
                                                   const _Float16* __restrict__ k,
                                                   const _Float16* __restrict__ v,
                                                   _Float16* __restrict__ o) {
  __shared__ __align__(16) _Float16 Kt[64 * 64];       // [key n][kd]
  __shared__ __align__(16) _Float16 Vs[64 * 64];       // [key kd][dim n]
  __shared__ __align__(16) _Float16 Ps[8 * 16 * 64];   // per-wave P tile [m][key]

  const int tid  = threadIdx.x;
  const int lane = tid & 31, wv = tid >> 5;
  const int n15  = lane & 15, g = lane >> 4;
  const int bh = blockIdx.x >> 4;
  const int qt = blockIdx.x & 15;
  const int b  = bh / HEADS, h = bh % HEADS;
  const int qrow = qt * 128 + wv * 16;

  const _Float16* qbase = q + ((size_t)(b * SEQ + qrow) * HEADS + h) * HD;
  const v16h qa0 = load_a_frag(qbase,      HEADS * HD, lane);
  const v16h qa1 = load_a_frag(qbase + 32, HEADS * HD, lane);

  // staging: 4 threads per key row, 16 halves each
  const int kr = tid >> 2, koff = (tid & 3) * 16;
  _Float16* lK = Kt + kr * 64 + koff;
  _Float16* lV = Vs + kr * 64 + koff;

  v8f oacc[4];
  float mrow[8], lrow[8];
#pragma unroll
  for (int t = 0; t < 4; ++t) oacc[t] = vzero8();
#pragma unroll
  for (int r = 0; r < 8; ++r) { mrow[r] = -3.0e38f; lrow[r] = 0.f; }

  for (int kc = 0; kc < SEQ; kc += 64) {
    const size_t src = ((size_t)(b * SEQ + kc + kr) * HEADS + h) * HD + koff;
    async_copy_b128(lK,     k + src);
    async_copy_b128(lK + 8, k + src + 8);
    async_copy_b128(lV,     v + src);
    async_copy_b128(lV + 8, v + src + 8);
    async_wait0();
    __syncthreads();

    // S = Q (16x64) @ K^T (64x64chunk)
    v8f sacc[4];
#pragma unroll
    for (int t = 0; t < 4; ++t) {
      sacc[t] = vzero8();
      v16h b0 = load_b_frag_t(Kt + (t * 16) * 64,      64, lane);
      sacc[t] = wmma_f16(qa0, b0, sacc[t]);
      v16h b1 = load_b_frag_t(Kt + (t * 16) * 64 + 32, 64, lane);
      sacc[t] = wmma_f16(qa1, b1, sacc[t]);
    }

    // online softmax (scale = 1/sqrt(64) = 0.125)
    _Float16* pw = Ps + wv * 1024;
#pragma unroll
    for (int r = 0; r < 8; ++r) {
      float s0 = sacc[0][r] * 0.125f, s1 = sacc[1][r] * 0.125f;
      float s2 = sacc[2][r] * 0.125f, s3 = sacc[3][r] * 0.125f;
      float mx = fmaxf(fmaxf(s0, s1), fmaxf(s2, s3));
#pragma unroll
      for (int off = 1; off < 16; off <<= 1) mx = fmaxf(mx, __shfl_xor(mx, off, 32));
      const float mn = fmaxf(mrow[r], mx);
      const float corr = __expf(mrow[r] - mn);
      mrow[r] = mn;
      float p0 = __expf(s0 - mn), p1 = __expf(s1 - mn);
      float p2 = __expf(s2 - mn), p3 = __expf(s3 - mn);
      float rs = p0 + p1 + p2 + p3;
#pragma unroll
      for (int off = 1; off < 16; off <<= 1) rs += __shfl_xor(rs, off, 32);
      lrow[r] = lrow[r] * corr + rs;
#pragma unroll
      for (int t = 0; t < 4; ++t) oacc[t][r] *= corr;
      const int prow = (r + 8 * g) * 64 + n15;
      pw[prow]      = (_Float16)p0;
      pw[prow + 16] = (_Float16)p1;
      pw[prow + 32] = (_Float16)p2;
      pw[prow + 48] = (_Float16)p3;
    }
    __syncthreads();

    // O += P (16x64) @ V (64x64)
    const v16h pa0 = load_a_frag(pw,      64, lane);
    const v16h pa1 = load_a_frag(pw + 32, 64, lane);
#pragma unroll
    for (int t = 0; t < 4; ++t) {
      v16h bv0 = load_b_frag_rm(Vs + t * 16,           64, lane);
      oacc[t] = wmma_f16(pa0, bv0, oacc[t]);
      v16h bv1 = load_b_frag_rm(Vs + 32 * 64 + t * 16, 64, lane);
      oacc[t] = wmma_f16(pa1, bv1, oacc[t]);
    }
    __syncthreads();
  }

  // write O / l  (f16, [row][h*64+d] for proj GEMM)
#pragma unroll
  for (int t = 0; t < 4; ++t)
#pragma unroll
    for (int r = 0; r < 8; ++r) {
      const int row = b * SEQ + qrow + r + 8 * g;
      const int col = h * HD + t * 16 + n15;
      o[(size_t)row * DIMC + col] = (_Float16)(oacc[t][r] / lrow[r]);
    }
}

// ---------------- host side ----------------
extern "C" void kernel_launch(void* const* d_in, const int* in_sizes, int n_in,
                              void* d_out, int out_size, void* d_ws, size_t ws_size,
                              hipStream_t stream) {
  const float* x      = (const float*)d_in[0];
  const float* ln1_w  = (const float*)d_in[1];
  const float* ln1_b  = (const float*)d_in[2];
  const float* qkv_w  = (const float*)d_in[3];
  const float* qkv_b  = (const float*)d_in[4];
  const float* proj_w = (const float*)d_in[5];
  const float* proj_b = (const float*)d_in[6];
  const float* ls1_g  = (const float*)d_in[7];
  const float* ln2_w  = (const float*)d_in[8];
  const float* ln2_b  = (const float*)d_in[9];
  const float* fc1_w  = (const float*)d_in[10];
  const float* fc1_b  = (const float*)d_in[11];
  const float* fc2_w  = (const float*)d_in[12];
  const float* fc2_b  = (const float*)d_in[13];
  const float* ls2_g  = (const float*)d_in[14];
  float* out = (float*)d_out;

  // workspace carve-up (256B aligned)
  char* p = (char*)d_ws;
  auto carve = [&](size_t bytes) { char* r = p; p += (bytes + 255) & ~(size_t)255; return r; };
  _Float16* ln1h   = (_Float16*)carve((size_t)ROWS * DIMC * 2);
  _Float16* qkvwt  = (_Float16*)carve((size_t)DIMC * 3 * DIMC * 2);   // [2304][768]
  _Float16* qh     = (_Float16*)carve((size_t)ROWS * DIMC * 2);
  _Float16* kh     = (_Float16*)carve((size_t)ROWS * DIMC * 2);
  _Float16* vh     = (_Float16*)carve((size_t)ROWS * DIMC * 2);
  _Float16* attnh  = (_Float16*)carve((size_t)ROWS * DIMC * 2);
  _Float16* projwt = (_Float16*)carve((size_t)DIMC * DIMC * 2);       // [768][768]
  float*    x1     = (float*)   carve((size_t)ROWS * DIMC * 4);
  _Float16* ln2h   = (_Float16*)carve((size_t)ROWS * DIMC * 2);
  _Float16* fc1wt  = (_Float16*)carve((size_t)DIMC * HIDDEN * 2);     // [3072][768]
  _Float16* h1     = (_Float16*)carve((size_t)ROWS * HIDDEN * 2);
  _Float16* fc2wt  = (_Float16*)carve((size_t)HIDDEN * DIMC * 2);     // [768][3072]
  (void)ws_size; (void)n_in; (void)in_sizes; (void)out_size;

  auto cvt_t = [&](const float* s, _Float16* d, int K, int Nn) {
    int n = K * Nn;
    cvt_t_kernel<<<(n + 255) / 256, 256, 0, stream>>>(s, d, K, Nn);
  };
  cvt_t(qkv_w, qkvwt, DIMC, 3 * DIMC);
  cvt_t(proj_w, projwt, DIMC, DIMC);
  cvt_t(fc1_w, fc1wt, DIMC, HIDDEN);
  cvt_t(fc2_w, fc2wt, HIDDEN, DIMC);

  // LN1
  ln_kernel<<<ROWS, 256, 0, stream>>>(x, ln1_w, ln1_b, ln1h);
  // QKV GEMM: 8192 x 2304 x 768
  gemm_kernel<0><<<dim3((3 * DIMC) / 128, ROWS / 128), 256, 0, stream>>>(
      ln1h, qkvwt, qkv_b, ROWS, 3 * DIMC, DIMC,
      nullptr, nullptr, nullptr, nullptr, qh, kh, vh);
  // attention
  attn_kernel<<<BATCH * HEADS * (SEQ / 128), 256, 0, stream>>>(qh, kh, vh, attnh);
  // proj GEMM + residual*ls1 -> x1
  gemm_kernel<1><<<dim3(DIMC / 128, ROWS / 128), 256, 0, stream>>>(
      attnh, projwt, proj_b, ROWS, DIMC, DIMC,
      x, ls1_g, x1, nullptr, nullptr, nullptr, nullptr);
  // LN2
  ln_kernel<<<ROWS, 256, 0, stream>>>(x1, ln2_w, ln2_b, ln2h);
  // FC1 + GELU -> h1 (f16)
  gemm_kernel<2><<<dim3(HIDDEN / 128, ROWS / 128), 256, 0, stream>>>(
      ln2h, fc1wt, fc1_b, ROWS, HIDDEN, DIMC,
      nullptr, nullptr, nullptr, h1, nullptr, nullptr, nullptr);
  // FC2 + residual*ls2 -> out
  gemm_kernel<1><<<dim3(DIMC / 128, ROWS / 128), 256, 0, stream>>>(
      h1, fc2wt, fc2_b, ROWS, DIMC, HIDDEN,
      x1, ls2_g, out, nullptr, nullptr, nullptr, nullptr);
}
